// LSTMClassifier_63711544869354
// MI455X (gfx1250) — matrix operand order repaired and checked by reading, not compiled, not run on hardware
//
#include <hip/hip_runtime.h>

typedef __attribute__((ext_vector_type(16))) __bf16 v16bf;
typedef __attribute__((ext_vector_type(8)))  __bf16 v8bf;
typedef __attribute__((ext_vector_type(8)))  float  v8f;

namespace {

constexpr int kS    = 512;   // sequence length
constexpr int kI    = 50;    // input features
constexpr int kH    = 64;    // hidden size
constexpr int kF1   = 32;    // MLP hidden
constexpr int kC    = 2;     // classes
constexpr int kG    = 4 * kH;   // 256 gate columns (i,f,g,o)
constexpr int kMT   = 16;       // batch rows per workgroup
constexpr int kKpad = 64;       // padded K for the input GEMM (50 -> 64)

__device__ __forceinline__ float fast_rcp(float v) {
  return __builtin_amdgcn_rcpf(v);     // bare v_rcp_f32, no IEEE-div fixup chain
}
__device__ __forceinline__ float fast_sigmoid(float v) {
  return fast_rcp(1.0f + __expf(-v));
}
__device__ __forceinline__ float fast_tanh(float v) {
#if __has_builtin(__builtin_amdgcn_tanhf)
  return __builtin_amdgcn_tanhf(v);    // gfx1250 v_tanh_f32
#elif __has_builtin(__builtin_amdgcn_tanh_f32)
  return __builtin_amdgcn_tanh_f32(v);
#else
  // 1 - 2/(e^{2v}+1): saturates correctly at +/-1 without inf/inf NaNs
  return 1.0f - 2.0f * fast_rcp(__expf(2.0f * v) + 1.0f);
#endif
}

// B-fragment for v_wmma_f32_16x16x32_bf16 (B is 32x16):
// element j, lanes 0-15: K=j (j<8), 8+j (j>=8) col=lane ; lanes 16-31: K=8+j (j<8), 16+j (j>=8) col=lane-16
__device__ __forceinline__ v16bf load_b_frag_whh(const float* __restrict__ W,
                                                 int colbase, int kstart, int lane) {
  const int col = colbase + (lane & 15);
  const int kb  = kstart + ((lane >> 4) << 3);
  const float* p = W + col * kH + kb;   // W_hh[col][k], contiguous in k
  v16bf r;
#pragma unroll
  for (int j = 0; j < 16; ++j) r[j] = (__bf16)p[(j < 8) ? j : j + 8];
  return r;
}

__device__ __forceinline__ v16bf load_b_frag_wih(const float* __restrict__ W,
                                                 int colbase, int kstart, int lane) {
  const int col = colbase + (lane & 15);
  const int kb  = kstart + ((lane >> 4) << 3);
  v16bf r;
#pragma unroll
  for (int j = 0; j < 16; ++j) {
    const int k = kb + ((j < 8) ? j : j + 8);
    r[j] = (k < kI) ? (__bf16)W[col * kI + k] : (__bf16)0.0f;  // zero-pad K 50..63
  }
  return r;
}

// A-fragment (16x32 bf16) from an LDS [16][64] bf16 tile:
// lanes 0-15 : row=lane, halves K kstart+0..7  then kstart+16..23
// lanes 16-31: row=lane-16, halves K kstart+8..15 then kstart+24..31
__device__ __forceinline__ v16bf load_a_frag(const __bf16* mat, int kstart, int lane) {
  const int row = lane & 15;
  const int off = (lane >> 4) << 3;
  const __bf16* p = mat + row * kKpad + kstart + off;
  const v8bf lo = *(const v8bf*)(p);        // ds_load_b128 (16B aligned)
  const v8bf hi = *(const v8bf*)(p + 16);   // ds_load_b128
  v16bf r;
#pragma unroll
  for (int j = 0; j < 8; ++j) { r[j] = lo[j]; r[8 + j] = hi[j]; }
  return r;
}

__device__ __forceinline__ v8f wmma_bf16(v16bf a, v16bf b, v8f c) {
  return __builtin_amdgcn_wmma_f32_16x16x32_bf16(
      /*neg_a=*/false, a, /*neg_b=*/false, b,
      /*c_mod=*/(short)0, c, /*reuse_a=*/false, /*reuse_b=*/false);
}

} // namespace

__launch_bounds__(256, 1)
__global__ void lstm_classifier_fused(const float* __restrict__ x,
                                      const float* __restrict__ W_ih,
                                      const float* __restrict__ W_hh,
                                      const float* __restrict__ b_ih,
                                      const float* __restrict__ b_hh,
                                      const float* __restrict__ W1,
                                      const float* __restrict__ b1,
                                      const float* __restrict__ W2,
                                      const float* __restrict__ b2,
                                      float* __restrict__ out) {
  __shared__ __align__(16) __bf16 x_bf[kMT * kKpad];   // current-step x tile (padded)
  __shared__ __align__(16) __bf16 h_bf[kMT * kH];      // h_{t-1} as bf16 A-operand
  __shared__ float g_sm[kMT * kG];                     // pre-activation gates, fp32
  __shared__ float h_f32[kMT * kH];                    // final h for the MLP head

  const int tid  = threadIdx.x;
  const int lane = tid & 31;
  const int wave = tid >> 5;
  const int b0   = blockIdx.x * kMT;

  // ---- resident weight/bias fragments (converted fp32 -> bf16 once) ----
  const int tile0 = wave * 2;
  const int tile1 = wave * 2 + 1;
  const int cb0 = tile0 * 16;
  const int cb1 = tile1 * 16;

  const v16bf wih00 = load_b_frag_wih(W_ih, cb0, 0,  lane);
  const v16bf wih01 = load_b_frag_wih(W_ih, cb0, 32, lane);
  const v16bf wih10 = load_b_frag_wih(W_ih, cb1, 0,  lane);
  const v16bf wih11 = load_b_frag_wih(W_ih, cb1, 32, lane);
  const v16bf whh00 = load_b_frag_whh(W_hh, cb0, 0,  lane);
  const v16bf whh01 = load_b_frag_whh(W_hh, cb0, 32, lane);
  const v16bf whh10 = load_b_frag_whh(W_hh, cb1, 0,  lane);
  const v16bf whh11 = load_b_frag_whh(W_hh, cb1, 32, lane);

  // bias fragment: C/D layout -> col = tilebase + (lane&15), same for all 8 rows
  const int ccol0 = cb0 + (lane & 15);
  const int ccol1 = cb1 + (lane & 15);
  const float bv0 = b_ih[ccol0] + b_hh[ccol0];
  const float bv1 = b_ih[ccol1] + b_hh[ccol1];
  v8f bias0, bias1;
#pragma unroll
  for (int r = 0; r < 8; ++r) { bias0[r] = bv0; bias1[r] = bv1; }

  // ---- per-thread cell state (4 elements each; never leaves registers) ----
  float c_reg[4]  = {0.f, 0.f, 0.f, 0.f};
  float h_keep[4] = {0.f, 0.f, 0.f, 0.f};

  // zero initial hidden state
#pragma unroll
  for (int rep = 0; rep < 4; ++rep) h_bf[tid + 256 * rep] = (__bf16)0.0f;
  __syncthreads();

  for (int t = 0; t < kS; ++t) {
    // ---- stage x[:, t, :] tile as padded bf16 in LDS ----
#pragma unroll
    for (int rep = 0; rep < 4; ++rep) {
      const int e   = tid + 256 * rep;       // 0..1023 over [16][64]
      const int row = e >> 6;
      const int i   = e & 63;
      const float v = (i < kI)
          ? x[((size_t)(b0 + row) * kS + t) * kI + i]
          : 0.0f;
      x_bf[e] = (__bf16)v;
    }
    // prefetch next timestep's slice while the dependent GEMM chain runs
    if (t + 1 < kS) {
      const int row = tid >> 4;              // 16 rows x 16 threads
      const int i   = (tid & 15) * 4;
      if (row < kMT && i < kI)
        __builtin_prefetch(&x[((size_t)(b0 + row) * kS + (t + 1)) * kI + i], 0, 3);
    }
    __syncthreads();

    // ---- gates GEMM: g = bias + x_t @ W_ih^T + h_{t-1} @ W_hh^T ----
    const v16bf xa0 = load_a_frag(x_bf, 0,  lane);
    const v16bf xa1 = load_a_frag(x_bf, 32, lane);
    const v16bf ha0 = load_a_frag(h_bf, 0,  lane);
    const v16bf ha1 = load_a_frag(h_bf, 32, lane);

    v8f acc0 = bias0;
    acc0 = wmma_bf16(xa0, wih00, acc0);
    acc0 = wmma_bf16(xa1, wih01, acc0);
    acc0 = wmma_bf16(ha0, whh00, acc0);
    acc0 = wmma_bf16(ha1, whh01, acc0);

    v8f acc1 = bias1;
    acc1 = wmma_bf16(xa0, wih10, acc1);
    acc1 = wmma_bf16(xa1, wih11, acc1);
    acc1 = wmma_bf16(ha0, whh10, acc1);
    acc1 = wmma_bf16(ha1, whh11, acc1);

    // C/D layout: VGPR r, lanes 0-15 -> M=r ; lanes 16-31 -> M=8+r
    {
      const int col_lo = lane & 15;
      const int rbase  = (lane >> 4) << 3;
#pragma unroll
      for (int r = 0; r < 8; ++r) {
        g_sm[(rbase + r) * kG + cb0 + col_lo] = acc0[r];
        g_sm[(rbase + r) * kG + cb1 + col_lo] = acc1[r];
      }
    }
    __syncthreads();

    // ---- elementwise LSTM cell update; c stays in registers ----
#pragma unroll
    for (int rep = 0; rep < 4; ++rep) {
      const int e   = tid + 256 * rep;
      const int row = e >> 6;
      const int hc  = e & 63;
      const float* gr = &g_sm[row * kG];
      const float ig = fast_sigmoid(gr[hc]);
      const float fg = fast_sigmoid(gr[kH + hc]);
      const float gg = fast_tanh(gr[2 * kH + hc]);
      const float og = fast_sigmoid(gr[3 * kH + hc]);
      const float cn = fg * c_reg[rep] + ig * gg;
      c_reg[rep] = cn;
      const float hn = og * fast_tanh(cn);
      h_keep[rep] = hn;
      h_bf[e] = (__bf16)hn;
    }
    __syncthreads();
  }

  // ---- publish final h (fp32) for the MLP head ----
#pragma unroll
  for (int rep = 0; rep < 4; ++rep) h_f32[tid + 256 * rep] = h_keep[rep];
  __syncthreads();

  // ---- head: out = relu(h @ W1^T + b1) @ W2^T + b2  (tiny: 32 threads) ----
  if (tid < kMT * kC) {
    const int row = tid >> 1;
    const int cc  = tid & 1;
    float acc = b2[cc];
#pragma unroll 4
    for (int f = 0; f < kF1; ++f) {
      float s = b1[f];
      const float* hr = &h_f32[row * kH];
      const float* wr = &W1[f * kH];
#pragma unroll
      for (int k = 0; k < kH; ++k) s += hr[k] * wr[k];
      s = fmaxf(s, 0.0f);
      acc += s * W2[cc * kF1 + f];
    }
    out[(b0 + row) * kC + cc] = acc;
  }
}

extern "C" void kernel_launch(void* const* d_in, const int* in_sizes, int n_in,
                              void* d_out, int out_size, void* d_ws, size_t ws_size,
                              hipStream_t stream) {
  const float* x    = (const float*)d_in[0];
  const float* W_ih = (const float*)d_in[1];
  const float* W_hh = (const float*)d_in[2];
  const float* b_ih = (const float*)d_in[3];
  const float* b_hh = (const float*)d_in[4];
  const float* W1   = (const float*)d_in[5];
  const float* b1   = (const float*)d_in[6];
  const float* W2   = (const float*)d_in[7];
  const float* b2   = (const float*)d_in[8];
  float* out = (float*)d_out;

  const int batch = in_sizes[0] / (kS * kI);   // 1024
  const int grid  = batch / kMT;               // 64 workgroups
  lstm_classifier_fused<<<grid, 256, 0, stream>>>(x, W_ih, W_hh, b_ih, b_hh,
                                                  W1, b1, W2, b2, out);
}
